// FCOSHead_37391985279626
// MI455X (gfx1250) — compile-verified, matrix operand order checked by reading, not run on hardware
//
#include <hip/hip_runtime.h>
#include <hip/hip_bf16.h>

typedef __attribute__((ext_vector_type(16))) _Float16 v16h;
typedef __attribute__((ext_vector_type(8)))  float    v8f;

#define NUM_B 32
#define NUM_P 21824
#define NUM_C 80
#define TOPK  1000
#define CAP   4096
#define NBIN  2048

__device__ __forceinline__ float sigmoidf(float x) { return 1.0f / (1.0f + __expf(-x)); }

// ---------------------------------------------------------------------------
// Kernel A: streaming score/argmax + WMMA box decode.
// grid = (ceil(P/256), B), block = 256 (8 waves of 32)
// ---------------------------------------------------------------------------
__global__ __launch_bounds__(256)
void fcos_score_decode(
    const float* __restrict__ c0, const float* __restrict__ c1, const float* __restrict__ c2,
    const float* __restrict__ c3, const float* __restrict__ c4,
    const float* __restrict__ e0, const float* __restrict__ e1, const float* __restrict__ e2,
    const float* __restrict__ e3, const float* __restrict__ e4,
    const float* __restrict__ r0, const float* __restrict__ r1, const float* __restrict__ r2,
    const float* __restrict__ r3, const float* __restrict__ r4,
    float* __restrict__ sc_ws, int* __restrict__ cl_ws, float* __restrict__ bx_ws)
{
    const int bb = blockIdx.y;
    int p = blockIdx.x * blockDim.x + threadIdx.x;
    if (p >= NUM_P) p = NUM_P - 1;   // clamp (dup writes, same value) -> EXEC all-ones for WMMA

    const float* clsP; const float* cenP; const float* regP;
    int hw, W, S, HW;
    if      (p < 16384) { clsP=c0; cenP=e0; regP=r0; hw=p;        W=128; S=8;   HW=16384; }
    else if (p < 20480) { clsP=c1; cenP=e1; regP=r1; hw=p-16384;  W=64;  S=16;  HW=4096;  }
    else if (p < 21504) { clsP=c2; cenP=e2; regP=r2; hw=p-20480;  W=32;  S=32;  HW=1024;  }
    else if (p < 21760) { clsP=c3; cenP=e3; regP=r3; hw=p-21504;  W=16;  S=64;  HW=256;   }
    else                { clsP=c4; cenP=e4; regP=r4; hw=p-21760;  W=8;   S=128; HW=64;    }

    const int yy = hw / W, xx = hw % W;
    const float cx = (float)(xx * S + (S >> 1));
    const float cy = (float)(yy * S + (S >> 1));

    // class max/argmax (first-max tie-break like jnp.argmax); coalesced across lanes per channel
    const long cbase = (long)bb * NUM_C * HW + hw;
    __builtin_prefetch(&clsP[cbase], 0, 0);
    float mx = clsP[cbase]; int mc = 0;
    for (int c = 1; c < NUM_C; ++c) {
        float v = clsP[cbase + (long)c * HW];
        if (v > mx) { mx = v; mc = c; }
    }
    const float cen = cenP[(long)bb * HW + hw];
    const long rbase = (long)bb * 4 * HW + hw;
    const float rl  = regP[rbase];
    const float rt  = regP[rbase + HW];
    const float rr  = regP[rbase + 2L * HW];
    const float rbo = regP[rbase + 3L * HW];

    const float score = sqrtf(sigmoidf(mx) * sigmoidf(cen));

    // --- box decode as WMMA affine transform: D = A(reg, f16 16x32) x B(+-1 diag, 32x16) + 0 ---
    const int lane = threadIdx.x & 31;
    const int wave = threadIdx.x >> 5;
    __shared__ float tile[8][32][4];

    // B-matrix: lane n (<16) holds K=halves; B[k][n] = -1 for k==n<2, +1 for k==n in {2,3}
    v16h bm = {};
    if (lane < 4) bm[lane] = (lane < 2) ? (_Float16)(-1.0f) : (_Float16)(1.0f);

    // A-matrix (anchors = lanes 0..15): lane m<16 holds K=0..7 in halves 0..7
    v16h a0 = {};
    if (lane < 16) { a0[0]=(_Float16)rl; a0[1]=(_Float16)rt; a0[2]=(_Float16)rr; a0[3]=(_Float16)rbo; }
    v8f cz = {};
    v8f d0 = __builtin_amdgcn_wmma_f32_16x16x32_f16(false, a0, false, bm, (short)0, cz, false, false);

    // second WMMA: anchors = lanes 16..31 (shuffle their regs into lanes 0..15)
    const float sl = __shfl(rl,  lane + 16, 32);
    const float st = __shfl(rt,  lane + 16, 32);
    const float sr = __shfl(rr,  lane + 16, 32);
    const float sB = __shfl(rbo, lane + 16, 32);
    v16h a1 = {};
    if (lane < 16) { a1[0]=(_Float16)sl; a1[1]=(_Float16)st; a1[2]=(_Float16)sr; a1[3]=(_Float16)sB; }
    v8f d1 = __builtin_amdgcn_wmma_f32_16x16x32_f16(false, a1, false, bm, (short)0, cz, false, false);

    // C/D layout: lane l holds N=l%16, rows M = r + (l<16 ? 0 : 8); only N<4 are real box coords
    const int n = lane & 15;
    if (n < 4) {
        const int mb = (lane < 16) ? 0 : 8;
        #pragma unroll
        for (int r2 = 0; r2 < 8; ++r2) {
            tile[wave][mb + r2][n]      = d0[r2];
            tile[wave][16 + mb + r2][n] = d1[r2];
        }
    }
    __syncthreads();
    const float b0 = cx + tile[wave][lane][0];
    const float b1 = cy + tile[wave][lane][1];
    const float b2 = cx + tile[wave][lane][2];
    const float b3 = cy + tile[wave][lane][3];

    const long o = (long)bb * NUM_P + p;
    sc_ws[o] = score;
    cl_ws[o] = mc + 1;
    bx_ws[o*4+0] = b0; bx_ws[o*4+1] = b1; bx_ws[o*4+2] = b2; bx_ws[o*4+3] = b3;
}

// ---------------------------------------------------------------------------
// Kernel B: per-batch exact top-1000 via LDS histogram-select + bitonic sort.
// grid = B, block = 256. Writes sc/cl/bx sections of d_out.
// ---------------------------------------------------------------------------
__global__ __launch_bounds__(256)
void fcos_topk(const float* __restrict__ sc_ws, const int* __restrict__ cl_ws,
               const float* __restrict__ bx_ws, float* __restrict__ out)
{
    const int bb  = blockIdx.x;
    const int tid = threadIdx.x;
    __shared__ unsigned int hist[NBIN];
    __shared__ unsigned long long cand[CAP];
    __shared__ unsigned int cnt, thr;

    for (int i = tid; i < NBIN; i += blockDim.x) hist[i] = 0u;
    if (tid == 0) cnt = 0u;
    __syncthreads();

    const long base = (long)bb * NUM_P;
    for (int p = tid; p < NUM_P; p += blockDim.x) {
        unsigned int key = __float_as_uint(sc_ws[base + p]);   // scores >= 0 -> monotone as uint
        atomicAdd(&hist[key >> 20], 1u);
    }
    __syncthreads();
    if (tid == 0) {
        unsigned int cum = 0, t = 0;
        for (int bin = NBIN - 1; bin >= 0; --bin) {
            cum += hist[bin];
            if (cum >= TOPK) { t = (unsigned int)bin; break; }
        }
        thr = t;
    }
    __syncthreads();
    const unsigned int T = thr;
    for (int p = tid; p < NUM_P; p += blockDim.x) {
        unsigned int key = __float_as_uint(sc_ws[base + p]);
        if ((key >> 20) >= T) {
            unsigned int pos = atomicAdd(&cnt, 1u);
            if (pos < CAP)
                cand[pos] = ((unsigned long long)key << 32)
                          | (unsigned long long)(0xFFFFFFFFu - (unsigned int)p); // desc score, asc idx
        }
    }
    __syncthreads();
    const unsigned int nc = (cnt > CAP) ? CAP : cnt;
    for (int i = tid; i < CAP; i += blockDim.x)
        if (i >= (int)nc) cand[i] = 0ull;
    __syncthreads();

    // bitonic sort, descending
    for (int k = 2; k <= CAP; k <<= 1) {
        for (int j = k >> 1; j > 0; j >>= 1) {
            for (int i = tid; i < CAP; i += blockDim.x) {
                const int ixj = i ^ j;
                if (ixj > i) {
                    unsigned long long x = cand[i], y = cand[ixj];
                    const bool desc = ((i & k) == 0);
                    if (desc ? (x < y) : (x > y)) { cand[i] = y; cand[ixj] = x; }
                }
            }
            __syncthreads();
        }
    }

    float* sc_o = out;
    float* cl_o = out + NUM_B * TOPK;
    float* bx_o = out + 2 * NUM_B * TOPK;
    for (int i = tid; i < TOPK; i += blockDim.x) {
        const unsigned long long kv = cand[i];
        const float s = __uint_as_float((unsigned int)(kv >> 32));
        unsigned int idx = 0xFFFFFFFFu - (unsigned int)(kv & 0xFFFFFFFFull);
        if (idx >= NUM_P) idx = NUM_P - 1;   // padding safety
        const long src = base + idx;
        sc_o[bb * TOPK + i] = s;
        cl_o[bb * TOPK + i] = (float)cl_ws[src];
        const long d = ((long)bb * TOPK + i) * 4;
        bx_o[d+0] = bx_ws[src*4+0];
        bx_o[d+1] = bx_ws[src*4+1];
        bx_o[d+2] = bx_ws[src*4+2];
        bx_o[d+3] = bx_ws[src*4+3];
    }
}

// ---------------------------------------------------------------------------
// Kernel C: per-batch class-aware greedy NMS fully in LDS. grid = B, block = 1024.
// ---------------------------------------------------------------------------
__global__ __launch_bounds__(1024)
void fcos_nms(float* __restrict__ out)
{
    const int bb  = blockIdx.x;
    const int tid = threadIdx.x;
    const float* sc_o = out;
    const float* cl_o = out + NUM_B * TOPK;
    const float* bx_o = out + 2 * NUM_B * TOPK;
    float*       kp_o = out + 6 * NUM_B * TOPK;

    __shared__ float sb[1024][4];
    __shared__ float ar[1024];
    __shared__ int   kp[1024];
    __shared__ float red[1024];

    float s = 0.f, cf = 0.f, x1 = 0.f, y1 = 0.f, x2 = 0.f, y2 = 0.f;
    if (tid < TOPK) {
        s  = sc_o[bb * TOPK + tid];
        cf = cl_o[bb * TOPK + tid];
        const long o = ((long)bb * TOPK + tid) * 4;
        x1 = bx_o[o]; y1 = bx_o[o+1]; x2 = bx_o[o+2]; y2 = bx_o[o+3];
    }
    // boxes.max() reduction over the batch's candidates
    red[tid] = (tid < TOPK) ? fmaxf(fmaxf(x1, y1), fmaxf(x2, y2)) : -3.4e38f;
    __syncthreads();
    for (int stp = 512; stp > 0; stp >>= 1) {
        if (tid < stp) red[tid] = fmaxf(red[tid], red[tid + stp]);
        __syncthreads();
    }
    const float off = (red[0] + 1.0f) * cf;       // class offset => no cross-class overlap
    sb[tid][0] = x1 + off; sb[tid][1] = y1 + off;
    sb[tid][2] = x2 + off; sb[tid][3] = y2 + off;
    ar[tid] = fmaxf(x2 - x1, 0.f) * fmaxf(y2 - y1, 0.f);
    kp[tid] = (tid < TOPK) && (s >= 0.05f);

    for (int i = 0; i < TOPK; ++i) {
        __syncthreads();                          // make iteration i-1 suppression visible
        if (kp[i]) {                              // uniform branch (same LDS value for all)
            const int j = tid;
            if (j > i && kp[j]) {
                const float ix1 = fmaxf(sb[i][0], sb[j][0]);
                const float iy1 = fmaxf(sb[i][1], sb[j][1]);
                const float ix2 = fminf(sb[i][2], sb[j][2]);
                const float iy2 = fminf(sb[i][3], sb[j][3]);
                const float iw = fmaxf(ix2 - ix1, 0.f);
                const float ih = fmaxf(iy2 - iy1, 0.f);
                const float inter = iw * ih;
                const float iou = inter / (ar[i] + ar[j] - inter + 1e-9f);
                if (iou > 0.6f) kp[j] = 0;
            }
        }
    }
    __syncthreads();
    if (tid < TOPK) kp_o[bb * TOPK + tid] = (float)kp[tid];
}

// ---------------------------------------------------------------------------
extern "C" void kernel_launch(void* const* d_in, const int* in_sizes, int n_in,
                              void* d_out, int out_size, void* d_ws, size_t ws_size,
                              hipStream_t stream)
{
    const float* cls[5]; const float* cen[5]; const float* reg[5];
    for (int i = 0; i < 5; ++i) {
        cls[i] = (const float*)d_in[i];
        cen[i] = (const float*)d_in[5 + i];
        reg[i] = (const float*)d_in[10 + i];
    }
    // workspace: B*P*(4 + 4 + 16) bytes ~= 16.8 MB
    float* sc_ws = (float*)d_ws;
    int*   cl_ws = (int*)(sc_ws + (size_t)NUM_B * NUM_P);
    float* bx_ws = (float*)(cl_ws + (size_t)NUM_B * NUM_P);
    float* out   = (float*)d_out;

    dim3 g1((NUM_P + 255) / 256, NUM_B);
    fcos_score_decode<<<g1, 256, 0, stream>>>(
        cls[0], cls[1], cls[2], cls[3], cls[4],
        cen[0], cen[1], cen[2], cen[3], cen[4],
        reg[0], reg[1], reg[2], reg[3], reg[4],
        sc_ws, cl_ws, bx_ws);
    fcos_topk<<<NUM_B, 256, 0, stream>>>(sc_ws, cl_ws, bx_ws, out);
    fcos_nms<<<NUM_B, 1024, 0, stream>>>(out);
}
